// SelectiveAttnTransformer_6236292514267
// MI455X (gfx1250) — matrix operand order repaired and checked by reading, not compiled, not run on hardware
//
#include <hip/hip_runtime.h>
#include <hip/hip_bf16.h>

// ---------------------------------------------------------------------------
// Model constants: V=512 D=256 H=8 E=32 W=200 SEG=16 B=2 L=8192 FF=1024
// lp=8200, nw=41, encoder n=512.
// Strategy: bf16 end-to-end for matrix data (weights pre-converted once per
// launch, activations produced in bf16), f32 residual stream, all GEMMs via
// v_wmma_f32_16x16x32_bf16. GEMM A-tiles staged with global_load_async_to_lds
// (ASYNCcnt) in a double-buffered pipeline so the DMA overlaps the WMMAs.
// ---------------------------------------------------------------------------

typedef __attribute__((ext_vector_type(16))) __bf16 v16bf;
typedef __attribute__((ext_vector_type(8)))  __bf16 v8bf;
typedef __attribute__((ext_vector_type(8)))  float  v8f;

union AFrag { v16bf v; v8bf h2[2]; };

__device__ __forceinline__ v8f wmma_bf16(v16bf a, v16bf b, v8f c) {
  return __builtin_amdgcn_wmma_f32_16x16x32_bf16(false, a, false, b, (short)0, c,
                                                 false, false);
}

__device__ __forceinline__ v8bf ld8(const __bf16* __restrict__ p, bool valid) {
  v8bf z;
#pragma unroll
  for (int i = 0; i < 8; ++i) z[i] = (__bf16)0.0f;
  return valid ? *(const v8bf*)p : z;
}

// ---------------------------------------------------------------------------
// Embedding + positional encodings (f32 residual stream).
// ---------------------------------------------------------------------------
__global__ __launch_bounds__(256) void embed_pe_kernel(
    const int* __restrict__ xi, const int* __restrict__ pos,
    const float* __restrict__ emb, float* __restrict__ h) {
  size_t idx = (size_t)blockIdx.x * 256 + threadIdx.x;   // B*L*256 exact
  int d = idx & 255;
  size_t r = idx >> 8;
  int l = (int)(r & 8191);
  int b = (int)(r >> 13);
  int tok = xi[(size_t)b * 8192 + l];
  float v = emb[(size_t)tok * 256 + d];
  int c = d, po = 0;
  if (c >= 128) { c -= 128; po = 1; }
  int p = pos[(size_t)b * 8193 + l + po];
  float pe;
  if (c < 5) {
    pe = (float)((p >> c) & 1);
  } else {
    int j = c - 5, i2 = j >> 1;
    float fr = __expf(-(float)(2 * i2) * (9.210340371976184f / 123.0f));
    float ang = (float)p * fr;
    pe = (j & 1) ? __cosf(ang) : __sinf(ang);
  }
  h[idx] = v + pe;
}

// ---------------------------------------------------------------------------
// Weight f32 -> bf16 conversion (once per launch, stream-ordered).
// ---------------------------------------------------------------------------
__global__ __launch_bounds__(256) void cvtw_kernel(const float* __restrict__ s,
                                                   __bf16* __restrict__ d,
                                                   int n) {
  int idx = blockIdx.x * 256 + threadIdx.x;
  if (idx < n) d[idx] = (__bf16)s[idx];
}

// ---------------------------------------------------------------------------
// Row LayerNorm over D=256, one wave/row; optional f32 and bf16 outputs.
// ---------------------------------------------------------------------------
__global__ __launch_bounds__(256) void layernorm_kernel(
    const float* __restrict__ x, const float* __restrict__ g,
    const float* __restrict__ bt, float* __restrict__ yf,
    __bf16* __restrict__ yb, int rows) {
  int lane = threadIdx.x & 31;
  int row = blockIdx.x * 8 + (threadIdx.x >> 5);
  if (row >= rows) return;
  const float* xr = x + (size_t)row * 256;
  float v[8]; float s = 0.f;
#pragma unroll
  for (int i = 0; i < 8; ++i) { v[i] = xr[lane + 32 * i]; s += v[i]; }
#pragma unroll
  for (int o = 16; o > 0; o >>= 1) s += __shfl_xor(s, o, 32);
  float mean = s * (1.0f / 256.0f);
  float q = 0.f;
#pragma unroll
  for (int i = 0; i < 8; ++i) { float d = v[i] - mean; q += d * d; }
#pragma unroll
  for (int o = 16; o > 0; o >>= 1) q += __shfl_xor(q, o, 32);
  float inv = rsqrtf(q * (1.0f / 256.0f) + 1e-5f);
#pragma unroll
  for (int i = 0; i < 8; ++i) {
    int c = lane + 32 * i;
    float o = (v[i] - mean) * inv * g[c] + bt[c];
    if (yf) yf[(size_t)row * 256 + c] = o;
    if (yb) yb[(size_t)row * 256 + c] = (__bf16)o;
  }
}

// ---------------------------------------------------------------------------
// bf16 WMMA GEMM: C[M,N] = act(A[M,K] @ W[K,N] + bias) (+ residual f32)
// A,W bf16; out f32 or bf16 (OUTBF). Block tile 128x128, K-step 32, 8 waves
// (4x2), each wave 32x64 = 2x4 WMMA. Double-buffered LDS pipeline: A tile
// kt+1 issued as async DMA (ASYNCcnt) and B tile kt+1 staged transposed
// through VGPRs while WMMAs consume tile kt; one barrier per K-step.
// M,N multiples of 128, K multiple of 32 (guaranteed by caller).
// ---------------------------------------------------------------------------
template <int ACT, bool BIAS, bool RES, bool OUTBF>
__global__ __launch_bounds__(256) void gemm_bf16_kernel(
    const __bf16* __restrict__ A, int lda,             // lda == K
    const __bf16* __restrict__ Bw, int ldb,            // [K][N], ldb == N
    const float* __restrict__ bias,                    // [N]
    const float* __restrict__ res,                     // [M][N] f32
    float* __restrict__ Cf, __bf16* __restrict__ Cb, int ldc) {
  __shared__ __align__(16) __bf16 As[2][128][40];   // [m][k], pad 32->40
  __shared__ __align__(16) __bf16 Bs[2][128][40];   // [n][k] transposed, pad
  int tid = threadIdx.x;
  int lane = tid & 31, wave = tid >> 5;
  int l16 = lane & 15, half = lane >> 4;
  int wM = wave >> 1, wN = wave & 1;
  int bm = blockIdx.y * 128, bn = blockIdx.x * 128;

  // staging coordinates (loop-invariant)
  int ar0 = (tid * 2) >> 2, ac0 = (tid * 2) & 3;         // A chunk 0
  int ar1 = (tid * 2 + 1) >> 2, ac1 = (tid * 2 + 1) & 3; // A chunk 1
  int bk = tid >> 3, bn_off = (tid & 7) * 16;            // B: k row, n chunk

  auto stageA = [&](int k0, int buf) {
    const __bf16* g0 = A + (size_t)(bm + ar0) * lda + k0 + ac0 * 8;
    unsigned l0 = (unsigned)(unsigned long long)(void*)&As[buf][ar0][ac0 * 8];
    asm volatile("global_load_async_to_lds_b128 %0, %1, off"
                 :: "v"(l0), "v"(g0) : "memory");
    const __bf16* g1 = A + (size_t)(bm + ar1) * lda + k0 + ac1 * 8;
    unsigned l1 = (unsigned)(unsigned long long)(void*)&As[buf][ar1][ac1 * 8];
    asm volatile("global_load_async_to_lds_b128 %0, %1, off"
                 :: "v"(l1), "v"(g1) : "memory");
  };
  auto stageB = [&](int k0, int buf) {
    const __bf16* src = Bw + (size_t)(k0 + bk) * ldb + bn + bn_off;
#pragma unroll
    for (int i = 0; i < 16; ++i) Bs[buf][bn_off + i][bk] = src[i];
  };

  v8f acc[2][4];
#pragma unroll
  for (int mi = 0; mi < 2; ++mi)
#pragma unroll
    for (int ni = 0; ni < 4; ++ni)
#pragma unroll
      for (int i = 0; i < 8; ++i) acc[mi][ni][i] = 0.f;

  int K = lda;
  int nIter = K >> 5;

  // prologue: stage tile 0 into buffer 0
  stageA(0, 0);
  stageB(0, 0);
  asm volatile("s_wait_asynccnt 0x0" ::: "memory");
  __syncthreads();

  for (int kt = 0; kt < nIter; ++kt) {
    int cur = kt & 1, nxt = cur ^ 1;
    bool hasNext = (kt + 1) < nIter;
    if (hasNext) {           // overlap: DMA + B staging for tile kt+1
      stageA((kt + 1) * 32, nxt);
      stageB((kt + 1) * 32, nxt);
    }

    AFrag aF[2], bF[4];
#pragma unroll
    for (int mi = 0; mi < 2; ++mi) {   // A pattern: k = 8h+e / 16+8h+e
      const __bf16* ar = &As[cur][wM * 32 + mi * 16 + l16][0];
      aF[mi].h2[0] = *(const v8bf*)(ar + 8 * half);
      aF[mi].h2[1] = *(const v8bf*)(ar + 16 + 8 * half);
    }
#pragma unroll
    for (int ni = 0; ni < 4; ++ni) {   // B pattern: k = 8h+e / 16+8h+e
      const __bf16* br = &Bs[cur][wN * 64 + ni * 16 + l16][0];
      bF[ni].h2[0] = *(const v8bf*)(br + 8 * half);
      bF[ni].h2[1] = *(const v8bf*)(br + 16 + 8 * half);
    }
#pragma unroll
    for (int mi = 0; mi < 2; ++mi)
#pragma unroll
      for (int ni = 0; ni < 4; ++ni)
        acc[mi][ni] = wmma_bf16(aF[mi].v, bF[ni].v, acc[mi][ni]);

    if (hasNext) {
      asm volatile("s_wait_asynccnt 0x0" ::: "memory");
      __syncthreads();
    }
  }

  // Epilogue: C VGPR r <-> (M = r + 8*half, N = l16). Fully specialized.
#pragma unroll
  for (int mi = 0; mi < 2; ++mi)
#pragma unroll
    for (int ni = 0; ni < 4; ++ni)
#pragma unroll
      for (int r = 0; r < 8; ++r) {
        int row = bm + wM * 32 + mi * 16 + half * 8 + r;
        int col = bn + wN * 64 + ni * 16 + l16;
        float vv = acc[mi][ni][r];
        if (BIAS) vv += bias[col];
        if (ACT == 1) vv = fmaxf(vv, 0.f);
        if (ACT == 2) vv = (vv > 0.f) ? vv : 0.01f * vv;
        if (RES) vv += res[(size_t)row * ldc + col];
        if (OUTBF) Cb[(size_t)row * ldc + col] = (__bf16)vv;
        else       Cf[(size_t)row * ldc + col] = vv;
      }
}

// ---------------------------------------------------------------------------
// Flash attention (WMMA QK^T and PV, online softmax), bf16 qkv in / bf16 out.
// One wave per 16-row q-tile; waves independent (wave-private LDS).
// mode 0: local windowed causal (W=200, keys contiguous [w*200-200, +399])
// mode 1: encoder full causal (n=512)
// qkv: [B*Ltok][768] bf16, q at col h*32, k at 256+h*32, v at 512+h*32.
// out: [B*Ltok][256] bf16 at col h*32.
// ---------------------------------------------------------------------------
__global__ __launch_bounds__(128) void flash_attn_kernel(
    const __bf16* __restrict__ qkv, __bf16* __restrict__ out,
    int Ltok, int mode, int totalTiles) {
  __shared__ __align__(16) float  sbuf[4][16][32];
  __shared__ __align__(16) __bf16 pbuf[4][16][32];
  __shared__ float alphaB[4][16];
  __shared__ float lB[4][16];

  int lane = threadIdx.x & 31;
  int wv = threadIdx.x >> 5;
  int tile = blockIdx.x * 4 + wv;
  if (tile >= totalTiles) return;

  int l16 = lane & 15, half = lane >> 4;
  int b, hh, qStart, t0, nSteps, rowCap;
  if (mode == 0) {
    const int TPW = 41 * 13;
    b = tile / (8 * TPW); int r0 = tile % (8 * TPW);
    hh = r0 / TPW; int t = r0 % TPW;
    int w = t / 13, qt = t % 13;
    qStart = w * 200 + qt * 16;
    t0 = w * 200 - 200;
    nSteps = 13;                    // 416 keys, tail masked
    rowCap = 200 - qt * 16;
  } else {
    b = tile / (8 * 32); int r0 = tile % (8 * 32);
    hh = r0 / 32; int qt = r0 % 32;
    qStart = qt * 16; t0 = 0; nSteps = (qt >> 1) + 1; rowCap = 16;
  }
  size_t bOfs = (size_t)b * Ltok;
  const float scale = 0.17677669529663687f;   // 1/sqrt(32)

  int tq_l = qStart + l16;
  bool qvld = tq_l < Ltok;
  const __bf16* qrow = qkv + (bOfs + (qvld ? tq_l : 0)) * 768 + hh * 32;
  AFrag qF;
  qF.h2[0] = ld8(qrow + 8 * half, qvld);
  qF.h2[1] = ld8(qrow + 16 + 8 * half, qvld);

  v8f o0, o1;
#pragma unroll
  for (int i = 0; i < 8; ++i) { o0[i] = 0.f; o1[i] = 0.f; }
  float mReg = -1e30f, lReg = 0.f;

  for (int s = 0; s < nSteps; ++s) {
    int kb = s * 32;
#pragma unroll
    for (int sub = 0; sub < 2; ++sub) {
      int tk = t0 + kb + sub * 16 + l16;
      bool kvld = (tk >= 0) && (tk < Ltok);
      const __bf16* krow = qkv + (bOfs + (kvld ? tk : 0)) * 768 + 256 + hh * 32;
      AFrag kF;                               // B pattern: k = 8h+e / 16+8h+e
      kF.h2[0] = ld8(krow + 8 * half, kvld);
      kF.h2[1] = ld8(krow + 16 + 8 * half, kvld);
      v8f zf;
#pragma unroll
      for (int i = 0; i < 8; ++i) zf[i] = 0.f;
      v8f sc = wmma_bf16(qF.v, kF.v, zf);
#pragma unroll
      for (int r = 0; r < 8; ++r)
        sbuf[wv][r + 8 * half][sub * 16 + l16] = sc[r];
    }
    asm volatile("s_wait_dscnt 0x0" ::: "memory");

    if (lane < 16) {
      int row = lane;
      int tq = qStart + row;
      float mx = mReg;
#pragma unroll
      for (int j = 0; j < 32; ++j) {
        int tk = t0 + kb + j;
        float xv = sbuf[wv][row][j] * scale;
        bool ok = (tk >= 0) && (tk < Ltok) && (tk <= tq);
        xv = ok ? xv : -1e30f;
        sbuf[wv][row][j] = xv;
        mx = fmaxf(mx, xv);
      }
      float alpha = __expf(mReg - mx);
      float sum = 0.f;
#pragma unroll
      for (int j = 0; j < 32; ++j) {
        float xv = sbuf[wv][row][j];
        float p = (xv > -1e29f) ? __expf(xv - mx) : 0.f;
        sum += p;
        pbuf[wv][row][j] = (__bf16)p;
      }
      mReg = mx;
      lReg = lReg * alpha + sum;
      alphaB[wv][row] = alpha;
    }
    asm volatile("s_wait_dscnt 0x0" ::: "memory");

#pragma unroll
    for (int r = 0; r < 8; ++r) {
      float a = alphaB[wv][r + 8 * half];
      o0[r] *= a; o1[r] *= a;
    }
    AFrag pF;
    const __bf16* pr = &pbuf[wv][l16][0];
    pF.h2[0] = *(const v8bf*)(pr + 8 * half);
    pF.h2[1] = *(const v8bf*)(pr + 16 + 8 * half);
#pragma unroll
    for (int nt = 0; nt < 2; ++nt) {
      int e = nt * 16 + l16;
      AFrag vF;
#pragma unroll
      for (int el = 0; el < 16; ++el) {
        int tk = t0 + kb + (el < 8 ? 8 * half + el : 16 + 8 * half + el - 8);    // B pattern: k = 8h+e / 16+8h+e
        bool ok = (tk >= 0) && (tk < Ltok);
        vF.v[el] = ok ? qkv[(bOfs + tk) * 768 + 512 + hh * 32 + e]
                      : (__bf16)0.0f;
      }
      if (nt == 0) o0 = wmma_bf16(pF.v, vF.v, o0);
      else         o1 = wmma_bf16(pF.v, vF.v, o1);
    }
  }

  if (lane < 16) lB[wv][lane] = lReg;
  asm volatile("s_wait_dscnt 0x0" ::: "memory");
#pragma unroll
  for (int r = 0; r < 8; ++r) {
    int row = r + 8 * half;
    int tq = qStart + row;
    if (row < rowCap && tq < Ltok) {
      float lsum = lB[wv][row];
      float inv = (lsum > 0.f) ? 1.0f / lsum : 0.f;
      __bf16* orow = out + (bOfs + tq) * 256 + hh * 32;
      orow[l16] = (__bf16)(o0[r] * inv);
      orow[16 + l16] = (__bf16)(o1[r] * inv);
    }
  }
}

// ---------------------------------------------------------------------------
// Elementwise helpers.
// ---------------------------------------------------------------------------
__global__ __launch_bounds__(256) void add_bf16_kernel(
    float* __restrict__ x, const __bf16* __restrict__ a) {
  size_t idx = (size_t)blockIdx.x * 256 + threadIdx.x;
  x[idx] += (float)a[idx];
}

__global__ __launch_bounds__(256) void leaky_cvt_kernel(
    const float* __restrict__ h, __bf16* __restrict__ o) {
  size_t idx = (size_t)blockIdx.x * 256 + threadIdx.x;
  float v = h[idx];
  o[idx] = (__bf16)((v > 0.f) ? v : 0.01f * v);
}

__global__ __launch_bounds__(256) void avgpool_kernel(
    const float* __restrict__ h, float* __restrict__ s,
    __bf16* __restrict__ sh) {
  int idx = blockIdx.x * 256 + threadIdx.x;   // 1024*256
  int d = idx & 255, r = idx >> 8;
  int i = r & 511, b = r >> 9;
  const float* base = h + ((size_t)b * 8192 + (size_t)i * 16) * 256 + d;
  float acc = 0.f;
#pragma unroll
  for (int j = 0; j < 16; ++j) acc += base[j * 256];
  acc *= (1.0f / 16.0f);
  s[(size_t)r * 256 + d] = acc;
  sh[(size_t)r * 256 + d] = (__bf16)acc;
}

__global__ __launch_bounds__(256) void upsample_add_kernel(
    const float* __restrict__ s, float* __restrict__ h) {
  size_t idx = (size_t)blockIdx.x * 256 + threadIdx.x;  // B*L*256
  int d = idx & 255;
  size_t r = idx >> 8;
  int l = (int)(r & 8191), b = (int)(r >> 13);
  float u = 0.f;
  if (l >= 15) {
    int i = (l - 15) >> 4;
    u = s[((size_t)b * 512 + i) * 256 + d];
  }
  h[idx] += u;
}

// ---------------------------------------------------------------------------
// Orchestration. Param flattening (dict insertion order):
// 0:x 1:pos 2:emb, in_blocks @3,@12 (qkv_w,w1,b1,w2,b2,n1g,n1b,n2g,n2b),
// enc @21,@33,@45,@57 (in_w,in_b,out_w,out_b,w1,b1,w2,b2,n1g,n1b,n2g,n2b),
// out_blocks @69,@78, ow=87, ob=88.
// ---------------------------------------------------------------------------
extern "C" void kernel_launch(void* const* d_in, const int* in_sizes, int n_in,
                              void* d_out, int out_size, void* d_ws,
                              size_t ws_size, hipStream_t stream) {
  (void)in_sizes; (void)n_in; (void)out_size; (void)ws_size;
  auto F = [&](int i) { return (const float*)d_in[i]; };
  const int* xi  = (const int*)d_in[0];
  const int* pos = (const int*)d_in[1];

  // ---- workspace carve: f32 region then bf16 region (~120 MB total) ----
  float* h    = (float*)d_ws;                         // [16384][256]
  float* sb   = h + (size_t)16384 * 256;              // [1024][256]
  float* stmp = sb + (size_t)1024 * 256;              // [1024][256]
  __bf16* lnT   = (__bf16*)(stmp + (size_t)1024 * 256);
  __bf16* qkv   = lnT   + (size_t)16384 * 256;
  __bf16* attn  = qkv   + (size_t)16384 * 768;
  __bf16* ffb   = attn  + (size_t)16384 * 256;
  __bf16* hbf   = ffb   + (size_t)16384 * 1024;
  __bf16* sbh   = hbf   + (size_t)16384 * 256;
  __bf16* sqkv  = sbh   + (size_t)1024 * 256;
  __bf16* sattn = sqkv  + (size_t)1024 * 768;
  __bf16* sff   = sattn + (size_t)1024 * 256;
  __bf16* wcur  = sff   + (size_t)1024 * 1024;

  // ---- convert all weights to bf16 (stream-ordered, before first GEMM) ----
  auto conv = [&](int pidx, int n) {
    __bf16* dst = wcur;
    cvtw_kernel<<<dim3((n + 255) / 256), dim3(256), 0, stream>>>(F(pidx), dst, n);
    wcur += n;
    return dst;
  };
  const int lb[4] = {3, 12, 69, 78};
  const int eb[4] = {21, 33, 45, 57};
  __bf16 *lw_qkv[4], *lw_w1[4], *lw_w2[4];
  for (int i = 0; i < 4; ++i) {
    lw_qkv[i] = conv(lb[i], 256 * 768);
    lw_w1[i]  = conv(lb[i] + 1, 256 * 1024);
    lw_w2[i]  = conv(lb[i] + 3, 1024 * 256);
  }
  __bf16 *ew_in[4], *ew_out[4], *ew_w1[4], *ew_w2[4];
  for (int i = 0; i < 4; ++i) {
    ew_in[i]  = conv(eb[i], 256 * 768);
    ew_out[i] = conv(eb[i] + 2, 256 * 256);
    ew_w1[i]  = conv(eb[i] + 4, 256 * 1024);
    ew_w2[i]  = conv(eb[i] + 6, 1024 * 256);
  }
  __bf16* owb = conv(87, 256 * 512);

  auto lnrm = [&](const float* xin, const float* g, const float* bt, float* yf,
                  __bf16* yb, int rows) {
    layernorm_kernel<<<dim3((rows + 7) / 8), dim3(256), 0, stream>>>(
        xin, g, bt, yf, yb, rows);
  };

  auto lmha = [&](int i) {
    int base = lb[i];
    lnrm(h, F(base + 5), F(base + 6), nullptr, lnT, 16384);
    gemm_bf16_kernel<0, false, false, true><<<dim3(6, 128), dim3(256), 0, stream>>>(
        lnT, 256, lw_qkv[i], 768, nullptr, nullptr, nullptr, qkv, 768);
    flash_attn_kernel<<<dim3(8528 / 4), dim3(128), 0, stream>>>(
        qkv, attn, 8192, 0, 8528);
    add_bf16_kernel<<<dim3(16384), dim3(256), 0, stream>>>(h, attn);
    lnrm(h, F(base + 7), F(base + 8), nullptr, lnT, 16384);
    gemm_bf16_kernel<2, true, false, true><<<dim3(8, 128), dim3(256), 0, stream>>>(
        lnT, 256, lw_w1[i], 1024, F(base + 2), nullptr, nullptr, ffb, 1024);
    gemm_bf16_kernel<0, true, true, false><<<dim3(2, 128), dim3(256), 0, stream>>>(
        ffb, 1024, lw_w2[i], 256, F(base + 4), h, h, nullptr, 256);
  };

  auto encl = [&](int i) {
    int base = eb[i];
    gemm_bf16_kernel<0, true, false, true><<<dim3(6, 8), dim3(256), 0, stream>>>(
        sbh, 256, ew_in[i], 768, F(base + 1), nullptr, nullptr, sqkv, 768);
    flash_attn_kernel<<<dim3(512 / 4), dim3(128), 0, stream>>>(
        sqkv, sattn, 512, 1, 512);
    gemm_bf16_kernel<0, true, true, false><<<dim3(2, 8), dim3(256), 0, stream>>>(
        sattn, 256, ew_out[i], 256, F(base + 3), sb, stmp, nullptr, 256);
    lnrm(stmp, F(base + 8), F(base + 9), sb, sbh, 1024);
    gemm_bf16_kernel<1, true, false, true><<<dim3(8, 8), dim3(256), 0, stream>>>(
        sbh, 256, ew_w1[i], 1024, F(base + 5), nullptr, nullptr, sff, 1024);
    gemm_bf16_kernel<0, true, true, false><<<dim3(2, 8), dim3(256), 0, stream>>>(
        sff, 1024, ew_w2[i], 256, F(base + 7), sb, stmp, nullptr, 256);
    lnrm(stmp, F(base + 10), F(base + 11), sb, sbh, 1024);
  };

  embed_pe_kernel<<<dim3(16384), dim3(256), 0, stream>>>(xi, pos, F(2), h);
  lmha(0);
  lmha(1);
  avgpool_kernel<<<dim3(1024), dim3(256), 0, stream>>>(h, sb, sbh);
  encl(0); encl(1); encl(2); encl(3);
  upsample_add_kernel<<<dim3(16384), dim3(256), 0, stream>>>(sb, h);
  lmha(2);
  lmha(3);
  leaky_cvt_kernel<<<dim3(16384), dim3(256), 0, stream>>>(h, hbf);
  gemm_bf16_kernel<0, true, false, false><<<dim3(4, 128), dim3(256), 0, stream>>>(
      hbf, 256, owb, 512, F(88), nullptr, (float*)d_out, nullptr, 512);
}
